// STAttention_11690900980414
// MI455X (gfx1250) — matrix operand order repaired and checked by reading, not compiled, not run on hardware
//
#include <hip/hip_runtime.h>
#include <hip/hip_bf16.h>

typedef __attribute__((ext_vector_type(16))) __bf16 v16bf;
typedef __attribute__((ext_vector_type(8)))  float  v8f;

#define B_TOT 8192
#define LE 40
#define LD 12
#define PQ 237
#define IN 256
#define H  64

// Padded LDS row strides (elements). Chosen so lane-stride of A-fragment
// loads is == 4 mod 64 DWORDs -> every ds_load_b128 hits a distinct 4-bank
// group (2-way, the floor for 128-bank-touch accesses) instead of 32-way.
// All strides keep rows 16-byte aligned for b128 vectorization.
#define P256 264   // 256-col arrays: 132 DW stride, l*132%64 = l*4
#define PHC  136   // 128-col [h|c]:   68 DW stride, l*68 %64 = l*4
#define P64   72   // 64-col arrays:   36 DW stride, distinct groups of 4

__device__ __forceinline__ float fast_tanh(float x){
  x = fminf(fmaxf(x, -15.f), 15.f);
  float e = __expf(2.f * x);
  return (e - 1.f) / (e + 1.f);
}
__device__ __forceinline__ float fast_sigm(float x){
  x = fminf(fmaxf(x, -30.f), 30.f);
  return 1.f / (1.f + __expf(-x));
}
__device__ __forceinline__ v8f vzero8(){
  v8f v;
  for (int i = 0; i < 8; i++) v[i] = 0.f;
  return v;
}

// A-matrix 16x32 bf16 fragment from row-major LDS tile (stride in elements).
// Lane l: M = l&15, half = l>>4; elems 0..7 -> K = k0 + 8*half + e,
// elems 8..15 -> K = k0 + 16 + 8*half + (e-8).   (ISA 7.12.2)
__device__ __forceinline__ v16bf load_a_frag(const __bf16* base, int stride,
                                             int m, int k0, int hl){
  const __bf16* p0 = base + m * stride + k0 + 8 * hl;
  v16bf a;
#pragma unroll
  for (int e = 0; e < 8; e++){ a[e] = p0[e]; a[8 + e] = p0[16 + e]; }
  return a;
}
// B-matrix 32x16 bf16 fragment: lane l holds N = n0 + (l&15); elems 0..7 -> K = k0 + 8*half + e,
// elems 8..15 -> K = k0 + 16 + 8*half + (e-8).
__device__ __forceinline__ v16bf load_b_frag(const __bf16* base, int stride,
                                             int k0, int n0, int lane){
  const __bf16* p = base + (k0 + 8 * (lane >> 4)) * stride + n0 + (lane & 15);
  v16bf b;
#pragma unroll
  for (int e = 0; e < 8; e++){ b[e] = p[e * stride]; b[8 + e] = p[(16 + e) * stride]; }
  return b;
}

#define WMMA_BF16(a, b, c) \
  __builtin_amdgcn_wmma_f32_16x16x32_bf16(false, (a), false, (b), (short)0, (c), false, false)

// ---------------------------------------------------------------------------
// Encoder: 4 waves/WG, each wave owns a 16-row batch tile for all 40 steps.
// All weights LDS-resident as bf16 W^T (bank-conflict-padded).  c-state lives
// in registers in WMMA C-fragment layout; h round-trips through LDS (it is
// next step's A-matrix).  ~313 KB of the 320 KB WGP LDS -> 1 WG/WGP, 4 waves
// filling the 4 SIMD32s.
// ---------------------------------------------------------------------------
__global__ __launch_bounds__(128, 1)
void enc_kernel(const float* __restrict__ ipq,
                const float* __restrict__ ehour, const float* __restrict__ ewday,
                const float* __restrict__ emon,  const float* __restrict__ ewoy,
                const float* __restrict__ Wi_w,  const float* __restrict__ Wi_b,
                const float* __restrict__ We_w,
                const float* __restrict__ Vd_w,  const float* __restrict__ Vd_b,
                const float* __restrict__ Wih,   const float* __restrict__ Whh,
                const float* __restrict__ bih,   const float* __restrict__ bhh,
                const int*   __restrict__ itime,
                __bf16*      __restrict__ mid)
{
  __shared__ __bf16 sWi [IN * P64];       // Wi_w^T  [k<256][n<64]
  __shared__ __bf16 sWe [2 * H * P64];    // We_w^T  [k<128][n<64]
  __shared__ __bf16 sVd [H * P256];       // Vd_w^T  [k<64 ][n<256]
  __shared__ __bf16 sWih[IN * P256];      // Wih^T   [k<256][n<256]
  __shared__ __bf16 sWhh[H * P256];       // Whh^T   [k<64 ][n<256]
  __shared__ float  sWib[H];
  __shared__ float  sVdb[IN];
  __shared__ float  sBih[4 * H];          // bih + bhh
  __shared__ __bf16 sX [4 * 16 * P256];   // per-wave x_t / x_in tile
  __shared__ __bf16 sHC[4 * 16 * PHC];    // per-wave [h|c] tile
  __shared__ __bf16 sS [4 * 16 * P64];    // per-wave tanh() tile

  const int tid  = threadIdx.x;
  const int lane = tid & 31;
  const int w    = tid >> 5;
  const int hl   = lane >> 4;     // lane half
  const int col  = lane & 15;     // C-layout N (mod 16) / A-layout M
  const int rowBase = blockIdx.x * 64 + w * 16;

  // Stage transposed weights into LDS (one-time; L2-friendly).
  for (int i = tid; i < IN * H;     i += 128){ int n = i & 63;  int k = i >> 6; sWi [k * P64  + n] = (__bf16)Wi_w[n * IN + k]; }
  for (int i = tid; i < 2 * H * H;  i += 128){ int n = i & 63;  int k = i >> 6; sWe [k * P64  + n] = (__bf16)We_w[n * 2 * H + k]; }
  for (int i = tid; i < H * IN;     i += 128){ int n = i & 255; int k = i >> 8; sVd [k * P256 + n] = (__bf16)Vd_w[n * H + k]; }
  for (int i = tid; i < IN * 4 * H; i += 128){ int n = i & 255; int k = i >> 8; sWih[k * P256 + n] = (__bf16)Wih [n * IN + k]; }
  for (int i = tid; i < H * 4 * H;  i += 128){ int n = i & 255; int k = i >> 8; sWhh[k * P256 + n] = (__bf16)Whh [n * H + k]; }
  for (int i = tid; i < H;      i += 128) sWib[i] = Wi_b[i];
  for (int i = tid; i < IN;     i += 128) sVdb[i] = Vd_b[i];
  for (int i = tid; i < 4 * H;  i += 128) sBih[i] = bih[i] + bhh[i];
  for (int i = tid; i < 4 * 16 * PHC; i += 128) sHC[i] = (__bf16)0.f;
  __syncthreads();

  __bf16* sx  = sX  + w * 16 * P256;
  __bf16* shc = sHC + w * 16 * PHC;
  __bf16* ss  = sS  + w * 16 * P64;

  v8f c_st[4];
#pragma unroll
  for (int j = 0; j < 4; j++) c_st[j] = vzero8();

#pragma unroll 1
  for (int t = 0; t < LE; t++){
    // ---- stage x_t = [input_p_q | embeddings] as bf16; prefetch t+1 ----
    for (int m = 0; m < 16; m++){
      const size_t b  = (size_t)(rowBase + m);
      const float* xp = ipq + (b * LE + t) * PQ;
      const int*   it = itime + (b * LE + t) * 4;
      for (int c = lane; c < IN; c += 32){
        float v;
        if      (c < PQ)      v = xp[c];
        else if (c < PQ + 5)  v = ehour[it[0] * 5 + (c - PQ)];
        else if (c < PQ + 8)  v = ewday[it[1] * 3 + (c - PQ - 5)];
        else if (c < PQ + 12) v = emon [it[2] * 4 + (c - PQ - 8)];
        else                  v = ewoy [it[3] * 7 + (c - PQ - 12)];
        sx[m * P256 + c] = (__bf16)v;
      }
      // global_prefetch next step's row (948 B = 4 lanes x 256 B lines)
      if (t + 1 < LE && lane < 4)
        __builtin_prefetch(xp + PQ + lane * 64, 0, 1);
    }
    __syncthreads();

    // ---- pre = x@Wi^T + [h,c]@We^T + Wi_b ; tanh -> sS ----
    v8f acc1[4];
#pragma unroll
    for (int j = 0; j < 4; j++) acc1[j] = vzero8();
#pragma unroll
    for (int k0 = 0; k0 < IN; k0 += 32){
      v16bf a = load_a_frag(sx, P256, col, k0, hl);
#pragma unroll
      for (int j = 0; j < 4; j++){
        v16bf bb = load_b_frag(sWi, P64, k0, 16 * j, lane);
        acc1[j] = WMMA_BF16(a, bb, acc1[j]);
      }
    }
#pragma unroll
    for (int k0 = 0; k0 < 2 * H; k0 += 32){
      v16bf a = load_a_frag(shc, PHC, col, k0, hl);
#pragma unroll
      for (int j = 0; j < 4; j++){
        v16bf bb = load_b_frag(sWe, P64, k0, 16 * j, lane);
        acc1[j] = WMMA_BF16(a, bb, acc1[j]);
      }
    }
#pragma unroll
    for (int j = 0; j < 4; j++){
      float bia = sWib[16 * j + col];
#pragma unroll
      for (int r = 0; r < 8; r++){
        float v = fast_tanh(acc1[j][r] + bia);
        ss[(r + 8 * hl) * P64 + 16 * j + col] = (__bf16)v;
      }
    }
    __syncthreads();

    // ---- score = s@Vd^T + Vd_b ; softmax over IN=256 ; x_in = x * p ----
    v8f sc[16];
#pragma unroll
    for (int j = 0; j < 16; j++) sc[j] = vzero8();
#pragma unroll
    for (int k0 = 0; k0 < H; k0 += 32){
      v16bf a = load_a_frag(ss, P64, col, k0, hl);
#pragma unroll
      for (int j = 0; j < 16; j++){
        v16bf bb = load_b_frag(sVd, P256, k0, 16 * j, lane);
        sc[j] = WMMA_BF16(a, bb, sc[j]);
      }
    }
    float rs[8];
#pragma unroll
    for (int r = 0; r < 8; r++) rs[r] = 0.f;
#pragma unroll
    for (int j = 0; j < 16; j++){
      float bia = sVdb[16 * j + col];
#pragma unroll
      for (int r = 0; r < 8; r++){
        float e = __expf(sc[j][r] + bia);  // |score| <~ 3.3, no max-shift needed
        sc[j][r] = e;
        rs[r] += e;
      }
    }
#pragma unroll
    for (int r = 0; r < 8; r++){
      float s = rs[r];                 // row M = r + 8*hl lives in one 16-lane half
      s += __shfl_xor(s, 1, 32);
      s += __shfl_xor(s, 2, 32);
      s += __shfl_xor(s, 4, 32);
      s += __shfl_xor(s, 8, 32);
      rs[r] = 1.f / s;
    }
#pragma unroll
    for (int j = 0; j < 16; j++){
#pragma unroll
      for (int r = 0; r < 8; r++){
        int m = r + 8 * hl;
        int n = 16 * j + col;
        float p  = sc[j][r] * rs[r];
        float xv = (float)sx[m * P256 + n];
        sx[m * P256 + n] = (__bf16)(xv * p);   // in-place: x_t -> x_in
      }
    }
    __syncthreads();

    // ---- gates = x_in@Wih^T + h@Whh^T ----
    v8f g[16];
#pragma unroll
    for (int j = 0; j < 16; j++) g[j] = vzero8();
#pragma unroll
    for (int k0 = 0; k0 < IN; k0 += 32){
      v16bf a = load_a_frag(sx, P256, col, k0, hl);
#pragma unroll
      for (int j = 0; j < 16; j++){
        v16bf bb = load_b_frag(sWih, P256, k0, 16 * j, lane);
        g[j] = WMMA_BF16(a, bb, g[j]);
      }
    }
#pragma unroll
    for (int k0 = 0; k0 < H; k0 += 32){
      v16bf a = load_a_frag(shc, PHC, col, k0, hl);   // h = cols 0..63 of [h|c]
#pragma unroll
      for (int j = 0; j < 16; j++){
        v16bf bb = load_b_frag(sWhh, P256, k0, 16 * j, lane);
        g[j] = WMMA_BF16(a, bb, g[j]);
      }
    }

    // ---- LSTM pointwise in C-fragment layout; write h,c and mid ----
#pragma unroll
    for (int jj = 0; jj < 4; jj++){
#pragma unroll
      for (int r = 0; r < 8; r++){
        int n = 16 * jj + col;
        float gi = g[jj     ][r] + sBih[n];
        float gf = g[jj +  4][r] + sBih[H + n];
        float gc = g[jj +  8][r] + sBih[2 * H + n];
        float go = g[jj + 12][r] + sBih[3 * H + n];
        float cn = fast_sigm(gf) * c_st[jj][r] + fast_sigm(gi) * fast_tanh(gc);
        c_st[jj][r] = cn;
        float hv = fast_sigm(go) * fast_tanh(cn);
        int m = r + 8 * hl;
        shc[m * PHC + n]     = (__bf16)hv;
        shc[m * PHC + H + n] = (__bf16)cn;
        mid[((size_t)(rowBase + m) * LE + t) * H + n] = (__bf16)hv;
      }
    }
    __syncthreads();
  }
}

// ---------------------------------------------------------------------------
// Decoder: 1 GFLOP total -> VALU.  One block per 16 batch rows; mid & wd_mid
// staged in LDS; 12 sequential attention+LSTM(H=1) steps.
// ---------------------------------------------------------------------------
__global__ __launch_bounds__(128, 1)
void dec_kernel(const __bf16* __restrict__ mid,
                const float* __restrict__ label_p,
                const float* __restrict__ Wd_w,  const float* __restrict__ Wd_b,
                const float* __restrict__ Wd2_w,
                const float* __restrict__ Vdt_w, const float* __restrict__ Vdt_b,
                const float* __restrict__ dWih,  const float* __restrict__ dWhh,
                const float* __restrict__ dbih,  const float* __restrict__ dbhh,
                float* __restrict__ out)
{
  __shared__ __bf16 smid[16 * LE * H];   // 80 KB
  __shared__ __bf16 swdm[16 * LE * H];   // 80 KB
  __shared__ float  sWd[H * H];
  __shared__ float  sWdb[H];
  __shared__ float  sW2[H * 2];
  __shared__ float  sVt[H];
  __shared__ float  sq[16 * H];
  __shared__ float  sscore[16 * LE];
  __shared__ float  sctx[16 * H];
  __shared__ float  sh[16], sc2[16], spin[16];

  const int tid  = threadIdx.x;
  const int base = blockIdx.x * 16;

  for (int i = tid; i < H * H; i += 128) sWd[i] = Wd_w[i];
  for (int i = tid; i < H;     i += 128){ sWdb[i] = Wd_b[i]; sVt[i] = Vdt_w[i]; }
  for (int i = tid; i < 2 * H; i += 128) sW2[i] = Wd2_w[i];
  for (int i = tid; i < 16 * LE * H; i += 128) smid[i] = mid[(size_t)base * LE * H + i];
  if (tid < 16){
    sh[tid] = 0.f; sc2[tid] = 0.f;
    spin[tid] = label_p[(size_t)(base + tid) * LE + (LE - 1)];
  }
  __syncthreads();

  // wd_mid = mid @ Wd_w^T + Wd_b
  for (int idx = tid; idx < 16 * LE * H; idx += 128){
    int d  = idx & (H - 1);
    int rt = idx >> 6;
    const __bf16* mr = smid + rt * H;
    const float*  wr = sWd + d * H;
    float a = sWdb[d];
#pragma unroll 4
    for (int e = 0; e < H; e++) a += (float)mr[e] * wr[e];
    swdm[idx] = (__bf16)a;
  }
  __syncthreads();

  const float vtb = Vdt_b[0];
#pragma unroll 1
  for (int step = 0; step < LD; step++){
    // q = hi*Wd2[:,0] + ci*Wd2[:,1]
    for (int idx = tid; idx < 16 * H; idx += 128){
      int r = idx >> 6; int d = idx & (H - 1);
      sq[idx] = sh[r] * sW2[d * 2] + sc2[r] * sW2[d * 2 + 1];
    }
    __syncthreads();
    // score[b,t] = Vdt_b + sum_d tanh(q + wd_mid) * Vdt
    for (int idx = tid; idx < 16 * LE; idx += 128){
      int r = idx / LE; int t = idx - r * LE;
      const __bf16* wr = swdm + (r * LE + t) * H;
      const float*  qr = sq + r * H;
      float a = vtb;
#pragma unroll 4
      for (int d = 0; d < H; d++) a += fast_tanh(qr[d] + (float)wr[d]) * sVt[d];
      sscore[idx] = a;
    }
    __syncthreads();
    // ctx[b,d] = sum_t score * mid
    for (int idx = tid; idx < 16 * H; idx += 128){
      int r = idx >> 6; int d = idx & (H - 1);
      const __bf16* mr = smid + r * LE * H + d;
      const float* srow = sscore + r * LE;
      float a = 0.f;
#pragma unroll 4
      for (int t = 0; t < LE; t++) a += srow[t] * (float)mr[t * H];
      sctx[idx] = a;
    }
    __syncthreads();
    // H=1 LSTM cell per row
    if (tid < 16){
      int r = tid;
      float gg[4];
#pragma unroll
      for (int k = 0; k < 4; k++){
        const float* wr = dWih + k * (H + 1);
        float a = dbih[k] + dbhh[k] + sh[r] * dWhh[k];
        for (int j = 0; j < H; j++) a += sctx[r * H + j] * wr[j];
        a += spin[r] * wr[H];
        gg[k] = a;
      }
      float cn = fast_sigm(gg[1]) * sc2[r] + fast_sigm(gg[0]) * fast_tanh(gg[2]);
      float hn = fast_sigm(gg[3]) * fast_tanh(cn);
      sc2[r] = cn; sh[r] = hn;
      out[(size_t)(base + r) * LD + step] = hn;
    }
    __syncthreads();
  }
}

extern "C" void kernel_launch(void* const* d_in, const int* in_sizes, int n_in,
                              void* d_out, int out_size, void* d_ws, size_t ws_size,
                              hipStream_t stream) {
  const float* ipq    = (const float*)d_in[0];
  const float* labelp = (const float*)d_in[1];
  const float* ehour  = (const float*)d_in[2];
  const float* ewday  = (const float*)d_in[3];
  const float* emon   = (const float*)d_in[4];
  const float* ewoy   = (const float*)d_in[5];
  const float* Wi_w   = (const float*)d_in[6];
  const float* Wi_b   = (const float*)d_in[7];
  const float* We_w   = (const float*)d_in[8];
  const float* Vd_w   = (const float*)d_in[9];
  const float* Vd_b   = (const float*)d_in[10];
  const float* eWih   = (const float*)d_in[11];
  const float* eWhh   = (const float*)d_in[12];
  const float* ebih   = (const float*)d_in[13];
  const float* ebhh   = (const float*)d_in[14];
  const float* Wd_w   = (const float*)d_in[15];
  const float* Wd_b   = (const float*)d_in[16];
  const float* Wd2_w  = (const float*)d_in[17];
  const float* Vdt_w  = (const float*)d_in[18];
  const float* Vdt_b  = (const float*)d_in[19];
  const float* dWih   = (const float*)d_in[20];
  const float* dWhh   = (const float*)d_in[21];
  const float* dbih   = (const float*)d_in[22];
  const float* dbhh   = (const float*)d_in[23];
  const int*   itime  = (const int*)d_in[24];
  // d_in[25] = decoder_time (unused by the reference forward pass)

  __bf16* mid = (__bf16*)d_ws;     // B * LE * H bf16 = 40 MB scratch
  float*  out = (float*)d_out;     // (B, LD) f32

  enc_kernel<<<B_TOT / 64, 128, 0, stream>>>(ipq, ehour, ewday, emon, ewoy,
      Wi_w, Wi_b, We_w, Vd_w, Vd_b, eWih, eWhh, ebih, ebhh, itime, mid);
  dec_kernel<<<B_TOT / 16, 128, 0, stream>>>(mid, labelp, Wd_w, Wd_b, Wd2_w,
      Vdt_w, Vdt_b, dWih, dWhh, dbih, dbhh, out);
}
